// BipartiteSoftMatching_44375602102936
// MI455X (gfx1250) — matrix operand order repaired and checked by reading, not compiled, not run on hardware
//
#include <hip/hip_runtime.h>

// Problem constants (from reference): N=8, T=4096, C_METRIC=64, C_X=768.
#define N_B   8
#define T_TOK 4096
#define THALF 2048
#define CM    64
#define CX    768

typedef __attribute__((ext_vector_type(16))) _Float16 v16h;
typedef __attribute__((ext_vector_type(8)))  _Float16 v8h;
typedef __attribute__((ext_vector_type(2)))  _Float16 v2h;
typedef __attribute__((ext_vector_type(8)))  float    v8f;

#define AS1 __attribute__((address_space(1)))
#define AS3 __attribute__((address_space(3)))

// Exact pointee type of the async-LDS builtins (from hipcc diagnostic):
// 'int __attribute__((vector_size(8)))' in AS1 (global) / AS3 (LDS).
typedef int async_b64_t __attribute__((vector_size(8)));

#if __has_builtin(__builtin_amdgcn_global_load_async_to_lds_b64)
#define USE_ASYNC_LDS 1
#else
#define USE_ASYNC_LDS 0
#endif

static __device__ __forceinline__ void wait_asynccnt_1() {
#if __has_builtin(__builtin_amdgcn_s_wait_asynccnt)
    __builtin_amdgcn_s_wait_asynccnt(1);
#else
    asm volatile("s_wait_asynccnt 0x1" ::: "memory");
#endif
}
static __device__ __forceinline__ void wait_asynccnt_0() {
#if __has_builtin(__builtin_amdgcn_s_wait_asynccnt)
    __builtin_amdgcn_s_wait_asynccnt(0);
#else
    asm volatile("s_wait_asynccnt 0x0" ::: "memory");
#endif
}

// ---------------------------------------------------------------------------
// Kernel 1: L2-normalize each metric row, convert to f16, deinterleave into
// even-token (B-side) and odd-token (A-side) matrices. One wave per token.
// ---------------------------------------------------------------------------
__global__ void __launch_bounds__(256) k_normalize(const float* __restrict__ metric,
                                                   _Float16* __restrict__ mEven,
                                                   _Float16* __restrict__ mOdd) {
    const int  lane = threadIdx.x & 31;
    const long tok  = (((long)blockIdx.x * blockDim.x) + threadIdx.x) >> 5;
    if (tok >= (long)N_B * T_TOK) return;

    const float2 v = ((const float2*)(metric + tok * CM))[lane]; // 2 ch per lane
    float ss = v.x * v.x + v.y * v.y;
#pragma unroll
    for (int off = 16; off > 0; off >>= 1) ss += __shfl_xor(ss, off, 32);
    const float inv = rsqrtf(ss);

    const int  t = (int)(tok % T_TOK);
    const long n = tok / T_TOK;
    _Float16* dst = ((t & 1) ? mOdd : mEven) + ((n * THALF) + (t >> 1)) * CM;
    v2h h;
    h.x = (_Float16)(v.x * inv);
    h.y = (_Float16)(v.y * inv);
    ((v2h*)dst)[lane] = h;
}

// ---------------------------------------------------------------------------
// Kernel 2: bipartite match via WMMA.
//   scores = mOdd(A, MxK rows=j) x mEven^T(B, KxN cols=i), K=64 (2 x k32 WMMA)
//   argmax over j kept per lane (8 D rows/lane), folded across lane^16 at end.
// 8 waves/block, each wave owns one 16-wide i-tile. Odd-token chunks (16x64
// f16) are staged into LDS (rows padded to 144B for 16B alignment + bank
// spread), software-pipelined:
//   - async path: GLOBAL_LOAD_ASYNC_TO_LDS_B64 double-buffered, ASYNCcnt waits
//   - fallback:   VGPR-carried prefetch (load chunk k+1 issued before compute
//                 of chunk k; waited only at next iteration's ds_store)
// ---------------------------------------------------------------------------
#define ROWH 72   // padded LDS row length in halves (144 bytes)

__global__ void __launch_bounds__(256) k_match(const _Float16* __restrict__ mEven,
                                               const _Float16* __restrict__ mOdd,
                                               int* __restrict__ node_idx) {
#if USE_ASYNC_LDS
    __shared__ _Float16 lds[2 * 16 * ROWH];
#else
    __shared__ _Float16 lds[16 * ROWH];
#endif

    const int n      = blockIdx.x >> 4;   // batch
    const int g      = blockIdx.x & 15;   // i-tile group (8 tiles per block)
    const int wid    = threadIdx.x >> 5;
    const int lane   = threadIdx.x & 31;
    const int half   = lane >> 4;         // 0 or 1
    const int m      = lane & 15;
    const int i_tile = g * 8 + wid;

    // B fragment (32x16 f16): column N = m, halves hold K = half*16 + 0..15.
    const _Float16* erow = mEven + (((long)n * THALF) + i_tile * 16 + m) * CM;
    const v16h bf0 = *(const v16h*)(erow + half * 16);       // K 0..31
    const v16h bf1 = *(const v16h*)(erow + 32 + half * 16);  // K 32..63

    // Cooperative chunk staging: 16 rows x 128B, 8B per thread.
    const uint2* gsrc = (const uint2*)(mOdd + (long)n * THALF * CM);
    _Float16* lstore = lds + (threadIdx.x >> 4) * ROWH + (threadIdx.x & 15) * 4;

    float best = -3.4e38f;
    int   bidx = 0;

#if USE_ASYNC_LDS
    // (global AS1 src, lds AS3 dst, imm offset, imm cpol); pointers built via
    // integer casts: flat VA == AS1 VA; low 32 bits of a flat LDS VA == AS3
    // byte offset (aperture layout).
#define ASYNC_CHUNK(buf_, jb_)                                                   \
    __builtin_amdgcn_global_load_async_to_lds_b64(                               \
        (AS1 async_b64_t*)(uintptr_t)(gsrc + (long)(jb_) * 16 + threadIdx.x),    \
        (AS3 async_b64_t*)(uintptr_t)(uint32_t)(uintptr_t)(lstore +              \
                                                 (buf_) * (16 * ROWH)),          \
        0, 0)
    ASYNC_CHUNK(0, 0);
#else
    uint2 d = gsrc[threadIdx.x];          // chunk 0 in flight
#endif

    for (int jb = 0; jb < THALF; jb += 16) {
        const _Float16* arow;
#if USE_ASYNC_LDS
        const int cur = (jb >> 4) & 1;
        if (jb + 16 < THALF) {            // issue next chunk into other buffer
            ASYNC_CHUNK(cur ^ 1, jb + 16);
            wait_asynccnt_1();            // current chunk's DMA done (this wave)
        } else {
            wait_asynccnt_0();
        }
        __syncthreads();                  // all waves' pieces arrived
        arow = lds + cur * (16 * ROWH) + m * ROWH;
#else
        __syncthreads();                  // WAR: previous chunk's reads done
        *(uint2*)lstore = d;              // s_wait_loadcnt lands here only
        __syncthreads();                  // chunk visible to all waves
        if (jb + 16 < THALF)              // issue next; overlaps with compute
            d = gsrc[(long)(jb + 16) * 16 + threadIdx.x];
        arow = lds + m * ROWH;
#endif

        // A fragments (16x32 f16): halves 0..15 = K kb+half*16+0..15.
        union { v16h v; v8h h[2]; } a0, a1;
        a0.h[0] = *(const v8h*)(arow + half * 16);
        a0.h[1] = *(const v8h*)(arow + 8 + half * 16);
        a1.h[0] = *(const v8h*)(arow + 32 + half * 16);
        a1.h[1] = *(const v8h*)(arow + 40 + half * 16);

        v8f c = {};
        c = __builtin_amdgcn_wmma_f32_16x16x32_f16(false, a0.v, false, bf0,
                                                   (short)0, c, false, false);
        c = __builtin_amdgcn_wmma_f32_16x16x32_f16(false, a1.v, false, bf1,
                                                   (short)0, c, false, false);

        const int jrow = jb + half * 8;   // D rows: half 0 -> M=r, half 1 -> M=r+8
#pragma unroll
        for (int r = 0; r < 8; ++r) {
            const float s = c[r];
            if (s > best) { best = s; bidx = jrow + r; }
        }
#if USE_ASYNC_LDS
        __syncthreads();                  // reads done -> buffer reusable
#endif
    }

    // Fold the two lane-halves (each covers complementary j rows).
    const float ob = __shfl_xor(best, 16, 32);
    const int   oi = __shfl_xor(bidx, 16, 32);
    if (ob > best || (ob == best && oi < bidx)) { best = ob; bidx = oi; }
    if (half == 0)
        node_idx[((long)n * THALF) + i_tile * 16 + m] = bidx;
}

// ---------------------------------------------------------------------------
// Kernel 3: init odd output rows with x_odd; counts = 1. float4 granularity.
// ---------------------------------------------------------------------------
__global__ void __launch_bounds__(256) k_init(const float* __restrict__ x,
                                              float* __restrict__ out,
                                              float* __restrict__ counts) {
    const long idx = (long)blockIdx.x * blockDim.x + threadIdx.x;
    const long total = (long)N_B * THALF * (CX / 4);
    if (idx >= total) return;
    const long row = idx / (CX / 4);
    const int  c4  = (int)(idx % (CX / 4));
    const long n = row / THALF, j = row % THALF;
    const long base = ((n * T_TOK) + 2 * j + 1) * CX;
    ((float4*)(out + base))[c4] = ((const float4*)(x + base))[c4];
    if (c4 == 0) counts[row] = 1.0f;
}

// ---------------------------------------------------------------------------
// Kernel 4: scatter-add x_even rows into out[2*node_idx+1]; bump counts.
// ---------------------------------------------------------------------------
__global__ void __launch_bounds__(256) k_scatter(const float* __restrict__ x,
                                                 const int* __restrict__ node_idx,
                                                 float* __restrict__ out,
                                                 float* __restrict__ counts) {
    const long idx = (long)blockIdx.x * blockDim.x + threadIdx.x;
    const long total = (long)N_B * THALF * CX;
    if (idx >= total) return;
    const long row = idx / CX;           // (n, i) over even tokens
    const int  c   = (int)(idx % CX);
    const long n = row / THALF, i = row % THALF;
    const int  j = node_idx[row];
    const float v = x[((n * T_TOK) + 2 * i) * CX + c];
    atomicAdd(out + ((n * T_TOK) + 2 * j + 1) * CX + c, v);
    if (c == 0) atomicAdd(counts + n * THALF + j, 1.0f);
}

// ---------------------------------------------------------------------------
// Kernel 5: divide odd rows by counts (in place). float4 granularity.
// ---------------------------------------------------------------------------
__global__ void __launch_bounds__(256) k_divide(float* __restrict__ out,
                                                const float* __restrict__ counts) {
    const long idx = (long)blockIdx.x * blockDim.x + threadIdx.x;
    const long total = (long)N_B * THALF * (CX / 4);
    if (idx >= total) return;
    const long row = idx / (CX / 4);
    const int  c4  = (int)(idx % (CX / 4));
    const long n = row / THALF, j = row % THALF;
    const float inv = 1.0f / counts[row];
    float4* p = (float4*)(out + ((n * T_TOK) + 2 * j + 1) * CX);
    float4 v = p[c4];
    v.x *= inv; v.y *= inv; v.z *= inv; v.w *= inv;
    p[c4] = v;
}

// ---------------------------------------------------------------------------
// Kernel 6: gather even rows: out[2i] = out[2*node_idx[i]+1].
// ---------------------------------------------------------------------------
__global__ void __launch_bounds__(256) k_gather(const int* __restrict__ node_idx,
                                                float* __restrict__ out) {
    const long idx = (long)blockIdx.x * blockDim.x + threadIdx.x;
    const long total = (long)N_B * THALF * (CX / 4);
    if (idx >= total) return;
    const long row = idx / (CX / 4);
    const int  c4  = (int)(idx % (CX / 4));
    const long n = row / THALF, i = row % THALF;
    const int  j = node_idx[row];
    const float4* src = (const float4*)(out + ((n * T_TOK) + 2 * j + 1) * CX);
    float4*       dst = (float4*)(out + ((n * T_TOK) + 2 * i) * CX);
    dst[c4] = src[c4];
}

// ---------------------------------------------------------------------------
extern "C" void kernel_launch(void* const* d_in, const int* in_sizes, int n_in,
                              void* d_out, int out_size, void* d_ws, size_t ws_size,
                              hipStream_t stream) {
    const float* metric = (const float*)d_in[0];   // (8, 4096, 64)  f32
    const float* x      = (const float*)d_in[1];   // (8, 4096, 768) f32
    float* out = (float*)d_out;                    // (8, 4096, 768) f32

    // Workspace layout (~4.2 MB total):
    char* ws = (char*)d_ws;
    _Float16* mEven    = (_Float16*)(ws);                       // 2 MB
    _Float16* mOdd     = (_Float16*)(ws + (2u << 20));          // 2 MB
    int*      node_idx = (int*)     (ws + (4u << 20));          // 64 KB
    float*    counts   = (float*)   (ws + (4u << 20) + 65536);  // 64 KB

    // 1) normalize + f16 convert + even/odd split (one wave per token)
    {
        const long waves = (long)N_B * T_TOK;        // 32768
        k_normalize<<<(int)(waves / 8), 256, 0, stream>>>(metric, mEven, mOdd);
    }
    // 2) WMMA argmax match: 128 blocks x 8 waves, each wave one 16-wide i-tile
    k_match<<<N_B * 16, 256, 0, stream>>>(mEven, mOdd, node_idx);
    // 3) init odd rows + counts
    {
        const long total = (long)N_B * THALF * (CX / 4);
        k_init<<<(int)((total + 255) / 256), 256, 0, stream>>>(x, out, counts);
    }
    // 4) scatter-add even rows by node_idx
    {
        const long total = (long)N_B * THALF * CX;
        k_scatter<<<(int)((total + 255) / 256), 256, 0, stream>>>(x, node_idx, out, counts);
    }
    // 5) divide odd rows by counts
    {
        const long total = (long)N_B * THALF * (CX / 4);
        k_divide<<<(int)((total + 255) / 256), 256, 0, stream>>>(out, counts);
    }
    // 6) gather even rows from merged odd rows
    {
        const long total = (long)N_B * THALF * (CX / 4);
        k_gather<<<(int)((total + 255) / 256), 256, 0, stream>>>(node_idx, out);
    }
}
